// FluctuationAwareConv1d_2413771621069
// MI455X (gfx1250) — matrix operand order repaired and checked by reading, not compiled, not run on hardware
//
#include <hip/hip_runtime.h>
#include <hip/hip_bf16.h>

// Problem constants (from reference)
#define BB    32
#define CIN   128
#define COUT  256
#define KS    9
#define KEXP  8     // number of experts K
#define FD    64
#define LL    4096
#define PADW  4
#define KTOT  (KS * CIN)   // 1152 = GEMM K dimension

// WMMA vector types (gfx1250, wave32)
typedef __attribute__((ext_vector_type(16))) __bf16       v16bf;
typedef __attribute__((ext_vector_type(8)))  float        v8f;
typedef __attribute__((ext_vector_type(4)))  unsigned int v4u;

union Frag { v16bf v; v4u u[2]; };

static __device__ __forceinline__ unsigned short f32_to_bf16(float f) {
    unsigned int u = __builtin_bit_cast(unsigned int, f);
    // round-to-nearest-even
    u += 0x7FFFu + ((u >> 16) & 1u);
    return (unsigned short)(u >> 16);
}

// ---------------------------------------------------------------------------
// Kernel 1: routing MLP -> omega (B x K), one block per sample
// ---------------------------------------------------------------------------
__global__ void routing_kernel(const float* __restrict__ Z,
                               const float* __restrict__ ziw,   // (CIN, FD)
                               const float* __restrict__ zib,   // (CIN)
                               const float* __restrict__ w1,    // (CIN/2, CIN)
                               const float* __restrict__ b1,    // (CIN/2)
                               const float* __restrict__ w2,    // (K, CIN/2)
                               const float* __restrict__ b2,    // (K)
                               float* __restrict__ omega)       // (B, K)
{
    __shared__ float sP[CIN];
    __shared__ float sH[CIN / 2];
    __shared__ float sL[KEXP];
    const int b = blockIdx.x;
    const int t = threadIdx.x;          // 128 threads
    const float* Zb = Z + b * FD;

    // proj = Z @ z_in_w^T + b
    {
        float a = zib[t];
        #pragma unroll 4
        for (int f = 0; f < FD; ++f) a += Zb[f] * ziw[t * FD + f];
        sP[t] = a;
    }
    __syncthreads();
    if (t < CIN / 2) {
        float a = b1[t];
        #pragma unroll 4
        for (int c = 0; c < CIN; ++c) a += sP[c] * w1[t * CIN + c];
        sH[t] = fmaxf(a, 0.0f);
    }
    __syncthreads();
    if (t < KEXP) {
        float a = b2[t];
        #pragma unroll 4
        for (int j = 0; j < CIN / 2; ++j) a += sH[j] * w2[t * (CIN / 2) + j];
        sL[t] = a;
    }
    __syncthreads();
    if (t == 0) {
        float mx = sL[0];
        #pragma unroll
        for (int k = 1; k < KEXP; ++k) mx = fmaxf(mx, sL[k]);
        float e[KEXP]; float s = 0.0f;
        #pragma unroll
        for (int k = 0; k < KEXP; ++k) { e[k] = __expf(sL[k] - mx); s += e[k]; }
        const float inv = 1.0f / s;
        #pragma unroll
        for (int k = 0; k < KEXP; ++k) omega[b * KEXP + k] = e[k] * inv;
    }
}

// ---------------------------------------------------------------------------
// Kernel 2: b_tilde = omega @ bias_bank   (B x COUT)
// ---------------------------------------------------------------------------
__global__ void bias_mix_kernel(const float* __restrict__ omega,
                                const float* __restrict__ bias_bank, // (K, COUT)
                                float* __restrict__ btilde)          // (B, COUT)
{
    const int b  = blockIdx.x;
    const int co = threadIdx.x;         // 256 threads
    float a = 0.0f;
    #pragma unroll
    for (int k = 0; k < KEXP; ++k)
        a += omega[b * KEXP + k] * bias_bank[k * COUT + co];
    btilde[b * COUT + co] = a;
}

// ---------------------------------------------------------------------------
// Kernel 3: W_tilde -> bf16, re-laid out ks-major: Wt[b][co][ks*128+ci]
// grid = (9, 256, 32), block = 128 (thread = ci)
// ---------------------------------------------------------------------------
__global__ void weight_mix_kernel(const float* __restrict__ omega,
                                  const float* __restrict__ bank,  // (K,COUT,CIN,KS)
                                  unsigned short* __restrict__ Wt) // bf16 (B,COUT,KTOT)
{
    const int ks = blockIdx.x;
    const int co = blockIdx.y;
    const int b  = blockIdx.z;
    const int ci = threadIdx.x;
    float a = 0.0f;
    #pragma unroll
    for (int k = 0; k < KEXP; ++k)
        a += omega[b * KEXP + k] *
             bank[(((size_t)k * COUT + co) * CIN + ci) * KS + ks];
    Wt[((size_t)(b * COUT + co)) * KTOT + ks * CIN + ci] = f32_to_bf16(a);
}

// ---------------------------------------------------------------------------
// Kernel 4: implicit-GEMM conv via v_wmma_f32_16x16x32_bf16
// grid = (L/128, COUT/128, B), block = 256 (8 waves)
// block tile: 128 c_out x 128 l ; wave tile: 64 x 32 (4x2 WMMA frags)
// ---------------------------------------------------------------------------
#define LTILE 128
#define XS_L  (LTILE + 2 * PADW)   // 136 staged positions (halo = conv padding)
#define XS_S  136                  // row stride (shorts): 272B -> 16B aligned, no bank conflicts

__global__ void conv_wmma_kernel(const float* __restrict__ X,
                                 const unsigned short* __restrict__ Wt,
                                 const float* __restrict__ btilde,
                                 float* __restrict__ out)
{
    __shared__ __attribute__((aligned(16))) unsigned short Xs[XS_L * XS_S];

    const int b   = blockIdx.z;
    const int mt  = blockIdx.y;             // c_out half (0..1)
    const int l0  = blockIdx.x * LTILE;
    const int tid = threadIdx.x;

    // ---- stage X tile transposed into LDS: Xs[l_local][ci], bf16, zero halo ----
    const float* Xb = X + (size_t)b * CIN * LL;
    for (int idx = tid; idx < CIN * XS_L; idx += 256) {
        const int ci  = idx / XS_L;
        const int lxl = idx - ci * XS_L;
        const int gl  = l0 - PADW + lxl;
        float f = (gl >= 0 && gl < LL) ? Xb[(size_t)ci * LL + gl] : 0.0f;
        Xs[lxl * XS_S + ci] = f32_to_bf16(f);
    }
    __syncthreads();

    const int wave = tid >> 5;
    const int lane = tid & 31;
    const int wr   = wave & 1;              // m 64-half within block tile
    const int wc   = wave >> 1;             // n 32-slice (0..3)
    const int lo   = lane & 15;
    const int hi   = lane >> 4;             // 0 | 1

    const int cbase = mt * 128 + wr * 64;

    // per-m_sub A row base pointers (per-lane: row = cbase + m*16 + lo)
    const unsigned short* Arow[4];
    #pragma unroll
    for (int m = 0; m < 4; ++m)
        Arow[m] = Wt + ((size_t)(b * COUT + cbase + m * 16 + lo)) * KTOT;

    v8f acc[4][2];
    #pragma unroll
    for (int m = 0; m < 4; ++m)
        #pragma unroll
        for (int n = 0; n < 2; ++n)
            acc[m][n] = (v8f){0.f, 0.f, 0.f, 0.f, 0.f, 0.f, 0.f, 0.f};

    // ---- K loop: 9 taps x 4 chunks of 32 input channels ----
    for (int ks = 0; ks < KS; ++ks) {
        for (int cit = 0; cit < 4; ++cit) {
            const int k0 = ks * CIN + cit * 32;

            // B fragments from LDS (column-major via transposed staging)
            Frag Bf[2];
            #pragma unroll
            for (int n = 0; n < 2; ++n) {
                const int lxl  = wc * 32 + n * 16 + lo + ks;     // halo handles the shift
                const int base = lxl * XS_S + cit * 32 + hi * 16;
                Bf[n].u[0] = *(const v4u*)&Xs[base];
                Bf[n].u[1] = *(const v4u*)&Xs[base + 8];
            }
            // A fragments from global (L2-resident mixed weights)
            Frag Af[4];
            #pragma unroll
            for (int m = 0; m < 4; ++m) {
                const unsigned short* p = Arow[m] + k0 + hi * 16;
                Af[m].u[0] = *(const v4u*)p;
                Af[m].u[1] = *(const v4u*)(p + 8);
            }
            #pragma unroll
            for (int m = 0; m < 4; ++m)
                #pragma unroll
                for (int n = 0; n < 2; ++n)
                    acc[m][n] = __builtin_amdgcn_wmma_f32_16x16x32_bf16(
                        false, Af[m].v, false, Bf[n].v,
                        (short)0, acc[m][n], false, false);
        }
    }

    // ---- epilogue: add per-sample bias, store fp32 ----
    const float* bt = btilde + b * COUT;
    float* Ob = out + (size_t)b * COUT * LL;
    #pragma unroll
    for (int m = 0; m < 4; ++m) {
        #pragma unroll
        for (int n = 0; n < 2; ++n) {
            const int lcol = l0 + wc * 32 + n * 16 + lo;
            #pragma unroll
            for (int j = 0; j < 8; ++j) {
                const int co = cbase + m * 16 + hi * 8 + j;      // C/D layout: M = j + 8*hi
                Ob[(size_t)co * LL + lcol] = acc[m][n][j] + bt[co];
            }
        }
    }
}

// ---------------------------------------------------------------------------
// Host launcher
// ---------------------------------------------------------------------------
extern "C" void kernel_launch(void* const* d_in, const int* in_sizes, int n_in,
                              void* d_out, int out_size, void* d_ws, size_t ws_size,
                              hipStream_t stream)
{
    const float* X    = (const float*)d_in[0];
    const float* Z    = (const float*)d_in[1];
    const float* ziw  = (const float*)d_in[2];
    const float* zib  = (const float*)d_in[3];
    const float* w1   = (const float*)d_in[4];
    const float* b1   = (const float*)d_in[5];
    const float* w2   = (const float*)d_in[6];
    const float* b2   = (const float*)d_in[7];
    const float* bank = (const float*)d_in[8];
    const float* bias = (const float*)d_in[9];
    float* out = (float*)d_out;

    // workspace layout (needs ~19 MB):
    //   [0, 1KB)      omega   (B x K fp32)
    //   [1KB, 33KB)   b_tilde (B x COUT fp32)
    //   [64KB, +18MB) W_tilde (B x COUT x KTOT bf16, ks-major kappa)
    char* ws = (char*)d_ws;
    float*          omega  = (float*)ws;
    float*          btilde = (float*)(ws + 1024);
    unsigned short* Wt     = (unsigned short*)(ws + 65536);

    routing_kernel<<<BB, 128, 0, stream>>>(Z, ziw, zib, w1, b1, w2, b2, omega);
    bias_mix_kernel<<<BB, COUT, 0, stream>>>(omega, bias, btilde);
    weight_mix_kernel<<<dim3(KS, COUT, BB), CIN, 0, stream>>>(omega, bank, Wt);

    dim3 grid(LL / LTILE, COUT / 128, BB);   // (32, 2, 32)
    conv_wmma_kernel<<<grid, 256, 0, stream>>>(X, Wt, btilde, out);
}
